// NbeGNN_88639535055016
// MI455X (gfx1250) — compile-verified, matrix operand order checked by reading, not run on hardware
//
#include <hip/hip_runtime.h>

#define N_NODES 100000
#define N_EDGES 1600000
#define D_IN    128
#define D_HID   128
#define D_OUT   16
#define N_STRIPS (N_NODES / 16)   // 6250, exact

typedef __attribute__((ext_vector_type(16))) __bf16 v16bf;
typedef __attribute__((ext_vector_type(8)))  float  v8f;
typedef unsigned short ushort_t;

// ---------- helpers ----------
__device__ __forceinline__ ushort_t f2bf(float f) {
    unsigned int u = __float_as_uint(f);
    unsigned int r = u + 0x7FFFu + ((u >> 16) & 1u);   // round-to-nearest-even
    return (ushort_t)(r >> 16);
}

// Build a 16-element bf16 fragment from two contiguous 16B chunks.
// Matches CDNA5 16-bit A/B operand layout: per lane, VGPR0-3 hold 8
// consecutive K values, VGPR4-7 hold 8 consecutive K values 16 further on.
__device__ __forceinline__ v16bf load_frag(const ushort_t* p0, const ushort_t* p1) {
    union { uint4 u[2]; v16bf v; } t;
    t.u[0] = *(const uint4*)p0;
    t.u[1] = *(const uint4*)p1;
    return t.v;
}

// ---------- degree ----------
__global__ void k_init_deg(float* deg) {
    int i = blockIdx.x * blockDim.x + threadIdx.x;
    if (i < N_NODES) deg[i] = 1.0f;                     // self-loop
}
__global__ void k_scatter_deg(const int* __restrict__ dst, float* deg) {
    int e = blockIdx.x * blockDim.x + threadIdx.x;
    if (e < N_EDGES) atomicAdd(&deg[dst[e]], 1.0f);
}
__global__ void k_finish_deg(const float* __restrict__ deg,
                             float* __restrict__ dis, float* __restrict__ dinv) {
    int i = blockIdx.x * blockDim.x + threadIdx.x;
    if (i < N_NODES) {
        float d = deg[i];
        dis[i]  = rsqrtf(d);
        dinv[i] = 1.0f / d;
    }
}

// ---------- conversions / staging ----------
__global__ void k_cvt_bf16(const float* __restrict__ in, ushort_t* __restrict__ out, int n) {
    int i = blockIdx.x * blockDim.x + threadIdx.x;
    if (i < n) out[i] = f2bf(in[i]);
}
// Wt[n*128 + k] = bf16(W[k*128 + n])   (128x128)
__global__ void k_prep_w(const float* __restrict__ W, ushort_t* __restrict__ Wt) {
    int t = blockIdx.x * blockDim.x + threadIdx.x;
    if (t < 128 * 128) {
        int n = t >> 7, k = t & 127;
        Wt[t] = f2bf(W[k * 128 + n]);
    }
}
// Wrt[n*128 + k] = bf16(Wr[k*16 + n])  (16x128)
__global__ void k_prep_wr(const float* __restrict__ Wr, ushort_t* __restrict__ Wrt) {
    int t = blockIdx.x * blockDim.x + threadIdx.x;
    if (t < 16 * 128) {
        int n = t >> 7, k = t & 127;
        Wrt[t] = f2bf(Wr[k * 16 + n]);
    }
}
__global__ void k_zero(float* __restrict__ p, int n) {
    int i = blockIdx.x * blockDim.x + threadIdx.x;
    if (i < n) p[i] = 0.0f;
}

// ---------- WMMA GEMM: C[100000x128] = A_bf16[100000x128] @ W_bf16[128x128] ----------
// One wave per 16-row strip; 8 column tiles x 4 K-steps of v_wmma_f32_16x16x32_bf16.
__global__ void k_gemm_wmma(const ushort_t* __restrict__ Abf,
                            const ushort_t* __restrict__ Wt,
                            float* __restrict__ C) {
    int wave  = (blockIdx.x * blockDim.x + threadIdx.x) >> 5;
    int lane  = threadIdx.x & 31;
    if (wave >= N_STRIPS) return;                       // wave-uniform: EXEC stays full
    int m    = lane & 15;
    int half = lane >> 4;

    const ushort_t* arow = Abf + (size_t)(wave * 16 + m) * 128;
    v16bf a[4];
#pragma unroll
    for (int ks = 0; ks < 4; ++ks) {
        int kb = ks * 32 + half * 8;
        a[ks] = load_frag(arow + kb, arow + kb + 16);
    }

    int n = lane & 15;
#pragma unroll
    for (int nt = 0; nt < 8; ++nt) {
        const ushort_t* brow = Wt + (size_t)(nt * 16 + n) * 128;
        v8f c = {};
#pragma unroll
        for (int ks = 0; ks < 4; ++ks) {
            int kb = ks * 32 + half * 8;
            v16bf b = load_frag(brow + kb, brow + kb + 16);
            c = __builtin_amdgcn_wmma_f32_16x16x32_bf16(
                    false, a[ks], false, b, (short)0, c, false, false);
        }
        float* crow = C + (size_t)(wave * 16) * 128 + nt * 16 + n;
#pragma unroll
        for (int r = 0; r < 8; ++r)
            crow[(size_t)(r + half * 8) * 128] = c[r];
    }
}

// ---------- edge scatter: agg[dst] += h_t[src] * dis[src]*dis[dst] ----------
// 32 lanes per edge, float4 per lane: coalesced 512B gather + 4 f32 L2 atomics.
__global__ void k_scatter_edges(const int* __restrict__ src, const int* __restrict__ dst,
                                const float* __restrict__ dis,
                                const float* __restrict__ ht,
                                float* __restrict__ agg) {
    long long tid = (long long)blockIdx.x * blockDim.x + threadIdx.x;
    if (tid >= (long long)N_EDGES * 32) return;
    int e = (int)(tid >> 5);
    int c = (int)(tid & 31) << 2;
    int s = src[e], d = dst[e];
    float nrm = dis[s] * dis[d];
    const float4 v = *(const float4*)(ht + (size_t)s * 128 + c);
    float* ap = agg + (size_t)d * 128 + c;
    atomicAdd(ap + 0, v.x * nrm);
    atomicAdd(ap + 1, v.y * nrm);
    atomicAdd(ap + 2, v.z * nrm);
    atomicAdd(ap + 3, v.w * nrm);
}

// ---------- combine: h_bf = bf16(relu(agg + h_t*deg_inv + b)) ----------
__global__ void k_combine(const float* __restrict__ agg, const float* __restrict__ ht,
                          const float* __restrict__ dinv, const float* __restrict__ b,
                          ushort_t* __restrict__ hbf) {
    int i = blockIdx.x * blockDim.x + threadIdx.x;
    if (i < N_NODES * 128) {
        int node = i >> 7, ch = i & 127;
        float v = agg[i] + ht[i] * dinv[node] + b[ch];
        v = v > 0.0f ? v : 0.0f;
        hbf[i] = f2bf(v);
    }
}

// ---------- readout: out = sigmoid(h @ Wr + br)*0.8 + 0.1 (WMMA, 16 cols) ----------
__global__ void k_readout_wmma(const ushort_t* __restrict__ Abf,
                               const ushort_t* __restrict__ Wrt,
                               const float* __restrict__ br,
                               float* __restrict__ out) {
    int wave = (blockIdx.x * blockDim.x + threadIdx.x) >> 5;
    int lane = threadIdx.x & 31;
    if (wave >= N_STRIPS) return;
    int m    = lane & 15;
    int half = lane >> 4;
    int n    = lane & 15;

    const ushort_t* arow = Abf + (size_t)(wave * 16 + m) * 128;
    const ushort_t* brow = Wrt + (size_t)n * 128;
    v8f c = {};
#pragma unroll
    for (int ks = 0; ks < 4; ++ks) {
        int kb = ks * 32 + half * 8;
        v16bf a = load_frag(arow + kb, arow + kb + 16);
        v16bf b = load_frag(brow + kb, brow + kb + 16);
        c = __builtin_amdgcn_wmma_f32_16x16x32_bf16(
                false, a, false, b, (short)0, c, false, false);
    }
    float bias = br[n];
#pragma unroll
    for (int r = 0; r < 8; ++r) {
        int row = wave * 16 + r + half * 8;
        float x = c[r] + bias;
        float y = 0.8f / (1.0f + __expf(-x)) + 0.1f;
        out[(size_t)row * 16 + n] = y;
    }
}

// ---------- host orchestration ----------
extern "C" void kernel_launch(void* const* d_in, const int* in_sizes, int n_in,
                              void* d_out, int out_size, void* d_ws, size_t ws_size,
                              hipStream_t stream) {
    const float* x  = (const float*)d_in[0];
    const int*   ei = (const int*)d_in[1];
    const int*   src = ei;
    const int*   dst = ei + N_EDGES;
    const float* Wl[3] = { (const float*)d_in[2], (const float*)d_in[4], (const float*)d_in[6] };
    const float* bl[3] = { (const float*)d_in[3], (const float*)d_in[5], (const float*)d_in[7] };
    const float* Wr = (const float*)d_in[8];
    const float* br = (const float*)d_in[9];
    float* out = (float*)d_out;

    // workspace carve-up (256B aligned)
    char* ws = (char*)d_ws;
    size_t off = 0;
    auto carve = [&](size_t bytes) -> void* {
        void* p = ws + off;
        off += (bytes + 255) & ~(size_t)255;
        return p;
    };
    float*    deg  = (float*)   carve(sizeof(float) * N_NODES);
    float*    dis  = (float*)   carve(sizeof(float) * N_NODES);
    float*    dinv = (float*)   carve(sizeof(float) * N_NODES);
    float*    h_t  = (float*)   carve(sizeof(float) * (size_t)N_NODES * 128);
    float*    agg  = (float*)   carve(sizeof(float) * (size_t)N_NODES * 128);
    ushort_t* hbf  = (ushort_t*)carve(sizeof(ushort_t) * (size_t)N_NODES * 128);
    ushort_t* Wt   = (ushort_t*)carve(sizeof(ushort_t) * 128 * 128);
    ushort_t* Wrt  = (ushort_t*)carve(sizeof(ushort_t) * 16 * 128);
    (void)ws_size; (void)in_sizes; (void)n_in; (void)out_size;

    const int B = 256;
    const int gN    = (N_NODES + B - 1) / B;                 // node-wide
    const int gE    = (N_EDGES + B - 1) / B;                 // edge-wide
    const int gFeat = (N_NODES * 128 + B - 1) / B;           // feature-wide
    const int gEdge = (int)(((long long)N_EDGES * 32 + B - 1) / B);
    const int gGemm = (N_STRIPS + 7) / 8;                    // 8 waves / block

    // degrees
    k_init_deg   <<<gN, B, 0, stream>>>(deg);
    k_scatter_deg<<<gE, B, 0, stream>>>(dst, deg);
    k_finish_deg <<<gN, B, 0, stream>>>(deg, dis, dinv);

    // first-layer input in bf16
    k_cvt_bf16<<<gFeat, B, 0, stream>>>(x, hbf, N_NODES * 128);

    for (int l = 0; l < 3; ++l) {
        k_prep_w       <<<64, B, 0, stream>>>(Wl[l], Wt);
        k_gemm_wmma    <<<gGemm, B, 0, stream>>>(hbf, Wt, h_t);
        k_zero         <<<gFeat, B, 0, stream>>>(agg, N_NODES * 128);
        k_scatter_edges<<<gEdge, B, 0, stream>>>(src, dst, dis, h_t, agg);
        k_combine      <<<gFeat, B, 0, stream>>>(agg, h_t, dinv, bl[l], hbf);
    }

    k_prep_wr     <<<8, B, 0, stream>>>(Wr, Wrt);
    k_readout_wmma<<<gGemm, B, 0, stream>>>(hbf, Wrt, br, out);
}
